// Decoder_57681410785395
// MI455X (gfx1250) — matrix operand order repaired and checked by reading, not compiled, not run on hardware
//
#include <hip/hip_runtime.h>
#include <hip/hip_bf16.h>
#include <math.h>

typedef __bf16 bf16_t;
typedef __attribute__((ext_vector_type(16))) __bf16 v16bf;
typedef __attribute__((ext_vector_type(8))) float v8f;

#define WMMA_BF16(a, b, c) \
  __builtin_amdgcn_wmma_f32_16x16x32_bf16(false, (a), false, (b), (short)0, (c), false, false)

// ---------------- WMMA fragment loaders (contiguous k order, same in A and B) --------------------

// A-matrix 16x32 bf16 from LDS, row-major [M][K] with leading dim `ld`.
// lane L: m = L&15 ; kh = L>>4 ; VGPR j: K = kh*16+2j (16 contiguous bf16 per lane)
static __device__ inline v16bf frag_a_lds(const bf16_t* __restrict__ base, int ld, int lane) {
  const int m = lane & 15, kh = lane >> 4;
  const bf16_t* row = base + (size_t)m * ld;
  v16bf a;
#pragma unroll
  for (int j = 0; j < 4; ++j) {
    const int k0 = kh * 16 + 2 * j;
    a[2 * j] = row[k0];
    a[2 * j + 1] = row[k0 + 1];
    const int k1 = k0 + 8;
    a[8 + 2 * j] = row[k1];
    a[8 + 2 * j + 1] = row[k1 + 1];
  }
  return a;
}

// B-matrix 32x16 bf16 from LDS stored TRANSPOSED [N][K] with ld = 32.
// lane L: n = L&15 ; kh = L>>4 ; VGPR j: K = kh*16+2j  -> 16 contiguous bf16 per lane.
static __device__ inline v16bf frag_b_nk(const bf16_t* __restrict__ base, int lane) {
  const int n = lane & 15, kh = lane >> 4;
  const bf16_t* p = base + n * 32 + kh * 16;
  v16bf b;
#pragma unroll
  for (int j = 0; j < 16; ++j) b[j] = p[j];
  return b;
}

// ---------------- GEMM: Y[M,N] = X[M,K] @ W[K,N] (+R)(+ReLU) ------------------------------
// X fp32 row-major; Wt = bf16 weights pre-transposed to [N][K].
// Block tile 128x128x32; 8 waves (2x4); double-buffered LDS; W tile moved with
// GLOBAL_LOAD_ASYNC_TO_LDS_B128 (ASYNCcnt), X tile staged through VGPRs with f32->bf16 cvt.
static __device__ inline void async_w_tile(const bf16_t* __restrict__ Wt, bf16_t* dst,
                                           int bn, int k0, int K, int tid) {
  const int n = tid >> 1, kh = (tid & 1) * 16;  // 256 threads cover 128 x 32 bf16
  const bf16_t* g = Wt + (size_t)(bn + n) * K + k0 + kh;
  const unsigned lds = (unsigned)(size_t)(dst + n * 32 + kh);
  asm volatile(
      "global_load_async_to_lds_b128 %0, %1, off\n\t"
      "global_load_async_to_lds_b128 %0, %1, off offset:16"
      :: "v"(lds), "v"((unsigned long long)(size_t)g)
      : "memory");
}

__global__ __launch_bounds__(256) void gemm_bf16_wmma(
    const float* __restrict__ X, const bf16_t* __restrict__ Wt,
    const float* __restrict__ R, float* __restrict__ Y,
    int M, int N, int K, int relu) {
  __shared__ alignas(32) bf16_t Xs[2][128 * 32];
  __shared__ alignas(32) bf16_t Ws[2][128 * 32];
  const int tid = threadIdx.x, lane = tid & 31, wave = tid >> 5;
  const int wm = wave >> 2, wn = wave & 3;
  const int bm = blockIdx.y * 128, bn = blockIdx.x * 128;

  v8f acc[4][2] = {};
  float4 xr[4];

  auto load_x = [&](int k0) {
#pragma unroll
    for (int j = 0; j < 4; ++j) {
      const int i = tid * 4 + j * 1024;
      const int r = i >> 5, c = i & 31;
      xr[j] = *(const float4*)(X + (size_t)(bm + r) * K + k0 + c);
    }
  };
  auto store_x = [&](bf16_t* dst) {
#pragma unroll
    for (int j = 0; j < 4; ++j) {
      const int i = tid * 4 + j * 1024;
      const int r = i >> 5, c = i & 31;
      dst[r * 32 + c] = (bf16_t)xr[j].x;
      dst[r * 32 + c + 1] = (bf16_t)xr[j].y;
      dst[r * 32 + c + 2] = (bf16_t)xr[j].z;
      dst[r * 32 + c + 3] = (bf16_t)xr[j].w;
    }
  };

  // prologue: tile 0
  async_w_tile(Wt, Ws[0], bn, 0, K, tid);
  load_x(0);
  store_x(Xs[0]);
  asm volatile("s_wait_asynccnt 0" ::: "memory");
  __syncthreads();

  const int nsteps = K >> 5;
  for (int s = 0; s < nsteps; ++s) {
    const int cur = s & 1, nxt = cur ^ 1;
    const bool more = (s + 1) < nsteps;
    if (more) {                     // overlap next tile's data movement with WMMA
      async_w_tile(Wt, Ws[nxt], bn, (s + 1) * 32, K, tid);
      load_x((s + 1) * 32);
    }

    v16bf af[4], bfr[2];
#pragma unroll
    for (int mi = 0; mi < 4; ++mi)
      af[mi] = frag_a_lds(Xs[cur] + (wm * 64 + mi * 16) * 32, 32, lane);
#pragma unroll
    for (int ni = 0; ni < 2; ++ni)
      bfr[ni] = frag_b_nk(Ws[cur] + (wn * 32 + ni * 16) * 32, lane);
#pragma unroll
    for (int mi = 0; mi < 4; ++mi)
#pragma unroll
      for (int ni = 0; ni < 2; ++ni)
        acc[mi][ni] = WMMA_BF16(af[mi], bfr[ni], acc[mi][ni]);

    if (more) {
      store_x(Xs[nxt]);
      asm volatile("s_wait_asynccnt 0" ::: "memory");
    }
    __syncthreads();
  }

  // C/D layout: VGPR v, lane L -> M = v + 8*(L>>4), N = L&15
  const int nloc = lane & 15, mb = (lane >> 4) * 8;
#pragma unroll
  for (int mi = 0; mi < 4; ++mi) {
#pragma unroll
    for (int ni = 0; ni < 2; ++ni) {
      const int gn = bn + wn * 32 + ni * 16 + nloc;
#pragma unroll
      for (int v = 0; v < 8; ++v) {
        const int gm = bm + wm * 64 + mi * 16 + mb + v;
        float val = acc[mi][ni][v];
        if (R) val += R[(size_t)gm * N + gn];
        if (relu) val = fmaxf(val, 0.0f);
        Y[(size_t)gm * N + gn] = val;
      }
    }
  }
}

// ---------------- Flash-style attention with WMMA -----------------------------------------
// Q,K,V,ctx: [B, T|S, 512], head h in cols h*64..h*64+63. toks -> padding mask.
__global__ __launch_bounds__(256) void attn_wmma(
    const float* __restrict__ Q, const float* __restrict__ Kt,
    const float* __restrict__ Vt, const int* __restrict__ toks,
    float* __restrict__ ctx, int T, int S, int causal) {
  __shared__ alignas(32) bf16_t Ks[32 * 64];   // [s][d]  (QK^T B-frags read rows of s)
  __shared__ alignas(32) bf16_t VsT[64 * 32];  // [d][s]  (PV  B-frags read rows of d)
  __shared__ alignas(32) bf16_t Ps[8 * 16 * 32];
  __shared__ unsigned char smask[32];
  const int tid = threadIdx.x, lane = tid & 31, wave = tid >> 5;
  const int b = blockIdx.z, h = blockIdx.y;
  const int qbase = blockIdx.x * 128;
  const int qrow0 = qbase + wave * 16;
  const int D = 512;
  const float scale = 0.125f;  // 1/sqrt(64)

  // Q fragments for this wave's 16 rows (two K=32 steps over d=0..63)
  v16bf qa[2];
  {
    const int m = lane & 15, kh = lane >> 4;
    const float* row = Q + ((size_t)(b * T + qrow0 + m)) * D + h * 64;
#pragma unroll
    for (int ds = 0; ds < 2; ++ds) {
      v16bf a;
#pragma unroll
      for (int j = 0; j < 4; ++j) {
        const int k0 = ds * 32 + kh * 16 + 2 * j;
        a[2 * j] = (bf16_t)row[k0];
        a[2 * j + 1] = (bf16_t)row[k0 + 1];
        const int k1 = k0 + 8;
        a[8 + 2 * j] = (bf16_t)row[k1];
        a[8 + 2 * j + 1] = (bf16_t)row[k1 + 1];
      }
      qa[ds] = a;
    }
  }

  float mrow[8], lrow[8];
  v8f o[4] = {};
#pragma unroll
  for (int v = 0; v < 8; ++v) { mrow[v] = -3.0e38f; lrow[v] = 0.0f; }

  bf16_t* Pw = Ps + wave * 16 * 32;
  const int kh = lane >> 4, nloc = lane & 15;

  for (int s0 = 0; s0 < S; s0 += 32) {
    if (causal && s0 > qbase + 127) break;

    const float* Kb = Kt + ((size_t)(b * S + s0)) * D + h * 64;
    const float* Vb = Vt + ((size_t)(b * S + s0)) * D + h * 64;
#pragma unroll
    for (int i = tid * 4; i < 32 * 64; i += 1024) {
      const int r = i >> 6, c = i & 63;
      const float4 kv = *(const float4*)(Kb + (size_t)r * D + c);
      Ks[r * 64 + c] = (bf16_t)kv.x;
      Ks[r * 64 + c + 1] = (bf16_t)kv.y;
      Ks[r * 64 + c + 2] = (bf16_t)kv.z;
      Ks[r * 64 + c + 3] = (bf16_t)kv.w;
      const float4 vv = *(const float4*)(Vb + (size_t)r * D + c);
      VsT[(c + 0) * 32 + r] = (bf16_t)vv.x;  // transpose into [d][s]
      VsT[(c + 1) * 32 + r] = (bf16_t)vv.y;
      VsT[(c + 2) * 32 + r] = (bf16_t)vv.z;
      VsT[(c + 3) * 32 + r] = (bf16_t)vv.w;
    }
    if (tid < 32) smask[tid] = (toks[(size_t)b * S + s0 + tid] == 0) ? 1 : 0;
    __syncthreads();

    if (!(causal && s0 > qrow0 + 15)) {
      // scores: 16 q rows x 32 s cols ; B = Ks rows (n = s_local, contiguous k = d)
      v8f sc[2];
#pragma unroll
      for (int ns = 0; ns < 2; ++ns) {
        v8f c = {};
#pragma unroll
        for (int ds = 0; ds < 2; ++ds) {
          const bf16_t* base = Ks + (ns * 16) * 64 + ds * 32;
          const int n = lane & 15;
          const bf16_t* p = base + n * 64 + kh * 16;
          v16bf bk;
#pragma unroll
          for (int j = 0; j < 16; ++j) bk[j] = p[j];
          c = WMMA_BF16(qa[ds], bk, c);
        }
        sc[ns] = c;
      }

      float nmax[8];
#pragma unroll
      for (int v = 0; v < 8; ++v) nmax[v] = mrow[v];
#pragma unroll
      for (int ns = 0; ns < 2; ++ns) {
        const int sg = s0 + ns * 16 + nloc;
        const bool pad = smask[ns * 16 + nloc] != 0;
#pragma unroll
        for (int v = 0; v < 8; ++v) {
          const int qg = qrow0 + kh * 8 + v;
          float xv = sc[ns][v] * scale;
          if (pad || (causal && sg > qg)) xv = -1.0e9f;
          sc[ns][v] = xv;
          nmax[v] = fmaxf(nmax[v], xv);
        }
      }
#pragma unroll
      for (int v = 0; v < 8; ++v)
#pragma unroll
        for (int off = 1; off < 16; off <<= 1)
          nmax[v] = fmaxf(nmax[v], __shfl_xor(nmax[v], off, 32));

      float corr[8];
#pragma unroll
      for (int v = 0; v < 8; ++v) {
        corr[v] = __expf(mrow[v] - nmax[v]);
        mrow[v] = nmax[v];
      }
#pragma unroll
      for (int ns = 0; ns < 2; ++ns)
#pragma unroll
        for (int v = 0; v < 8; ++v) sc[ns][v] = __expf(sc[ns][v] - nmax[v]);
#pragma unroll
      for (int v = 0; v < 8; ++v) {
        float s = sc[0][v] + sc[1][v];
#pragma unroll
        for (int off = 1; off < 16; off <<= 1) s += __shfl_xor(s, off, 32);
        lrow[v] = lrow[v] * corr[v] + s;
      }
#pragma unroll
      for (int dsb = 0; dsb < 4; ++dsb)
#pragma unroll
        for (int v = 0; v < 8; ++v) o[dsb][v] *= corr[v];

      // stage P (C layout -> A layout) via per-wave LDS
#pragma unroll
      for (int ns = 0; ns < 2; ++ns)
#pragma unroll
        for (int v = 0; v < 8; ++v)
          Pw[(kh * 8 + v) * 32 + ns * 16 + nloc] = (bf16_t)sc[ns][v];
      v16bf pa = frag_a_lds(Pw, 32, lane);
#pragma unroll
      for (int dsb = 0; dsb < 4; ++dsb) {
        v16bf bv = frag_b_nk(VsT + (dsb * 16) * 32, lane);
        o[dsb] = WMMA_BF16(pa, bv, o[dsb]);
      }
    }
    __syncthreads();
  }

  float inv[8];
#pragma unroll
  for (int v = 0; v < 8; ++v) inv[v] = 1.0f / fmaxf(lrow[v], 1e-20f);
  float* Cb = ctx + ((size_t)(b * T + qrow0)) * D + h * 64;
#pragma unroll
  for (int dsb = 0; dsb < 4; ++dsb)
#pragma unroll
    for (int v = 0; v < 8; ++v)
      Cb[(size_t)(kh * 8 + v) * D + dsb * 16 + nloc] = o[dsb][v] * inv[v];
}

// ---------------- Weight f32 [K][N] -> bf16 transposed [N][K] (LDS-tiled) ----------------
__global__ __launch_bounds__(256) void transpose_cvt_bf16(
    const float* __restrict__ in, bf16_t* __restrict__ out, int K, int N) {
  __shared__ float t[32][33];
  const int k0 = blockIdx.y * 32, n0 = blockIdx.x * 32;
  const int tx = threadIdx.x & 31, ty = threadIdx.x >> 5;
#pragma unroll
  for (int i = ty; i < 32; i += 8) t[i][tx] = in[(size_t)(k0 + i) * N + n0 + tx];
  __syncthreads();
#pragma unroll
  for (int i = ty; i < 32; i += 8)
    out[(size_t)(n0 + i) * K + k0 + tx] = (bf16_t)t[tx][i];
}

// ---------------- LayerNorm over last dim 512, one wave32 per row -------------------------
__global__ __launch_bounds__(256) void layernorm_rows(float* __restrict__ X, int rows) {
  const int row = blockIdx.x * 8 + (threadIdx.x >> 5);
  const int lane = threadIdx.x & 31;
  if (row >= rows) return;
  float* p = X + (size_t)row * 512;
  float vals[16], s = 0.0f, ss = 0.0f;
#pragma unroll
  for (int i = 0; i < 16; ++i) {
    const float v = p[lane + i * 32];
    vals[i] = v;
    s += v;
    ss += v * v;
  }
#pragma unroll
  for (int off = 1; off < 32; off <<= 1) {
    s += __shfl_xor(s, off, 32);
    ss += __shfl_xor(ss, off, 32);
  }
  const float mu = s * (1.0f / 512.0f);
  const float var = ss * (1.0f / 512.0f) - mu * mu;
  const float r = rsqrtf(var + 1e-5f);
#pragma unroll
  for (int i = 0; i < 16; ++i) p[lane + i * 32] = (vals[i] - mu) * r;
}

// ---------------- Embedding (padding_idx=0) + sinusoidal PE -------------------------------
__global__ __launch_bounds__(256) void embed_pe(
    const int* __restrict__ toks, const float* __restrict__ emb,
    float* __restrict__ x, int T) {
  const size_t idx = (size_t)blockIdx.x * 256 + threadIdx.x;  // over B*T*512
  const int d = (int)(idx & 511);
  const size_t bt = idx >> 9;
  const int t = (int)(bt % (size_t)T);
  const int tok = toks[bt];
  const float e = (tok == 0) ? 0.0f : emb[(size_t)tok * 512 + d];
  const int d2 = d & ~1;
  const float div = __expf(-0.01798894604f * (float)d2);  // ln(10000)/512
  const float ang = (float)t * div;
  const float pe = (d & 1) ? __cosf(ang) : __sinf(ang);
  x[idx] = e + pe;
}

// ---------------- Host orchestration ------------------------------------------------------
extern "C" void kernel_launch(void* const* d_in, const int* in_sizes, int n_in,
                              void* d_out, int out_size, void* d_ws, size_t ws_size,
                              hipStream_t stream) {
  (void)in_sizes; (void)n_in; (void)out_size; (void)ws_size;
  const int B = 8, T = 1024, S = 1024, D = 512, H = 8, FF = 2048, L = 6;
  const int M = B * T;  // == B * S

  const int* dec = (const int*)d_in[0];
  const int* enc = (const int*)d_in[1];
  const float* enc_out = (const float*)d_in[2];
  const float* emb = (const float*)d_in[3];
  const float* Wsrc[8] = {(const float*)d_in[4], (const float*)d_in[5],
                          (const float*)d_in[6], (const float*)d_in[7],
                          (const float*)d_in[8], (const float*)d_in[9],
                          (const float*)d_in[10], (const float*)d_in[11]};
  const float* W1 = (const float*)d_in[12];
  const float* W2 = (const float*)d_in[13];

  float* x = (float*)d_out;  // live activation [B,T,512]
  float* Qb = (float*)d_ws;
  float* Kb = Qb + (size_t)M * D;
  float* Vb = Kb + (size_t)M * D;
  float* Cb = Vb + (size_t)M * D;
  float* Hb = Cb + (size_t)M * D;                   // [M, FF]
  bf16_t* Wt = (bf16_t*)(Hb + (size_t)M * FF);      // bf16 transposed weights
  const size_t PROJ = (size_t)D * D;
  const size_t FFE = (size_t)D * FF;
  const size_t LSTR = 8 * PROJ + 2 * FFE;           // elems per layer

  // one-time (per call) weight pre-conversion: f32 [K][N] -> bf16 [N][K]
  {
    const dim3 blk(256);
    const dim3 gDD(D / 32, D / 32), gDF(FF / 32, D / 32), gFD(D / 32, FF / 32);
    for (int l = 0; l < L; ++l) {
      bf16_t* base = Wt + (size_t)l * LSTR;
      for (int p = 0; p < 8; ++p)
        transpose_cvt_bf16<<<gDD, blk, 0, stream>>>(Wsrc[p] + (size_t)l * PROJ,
                                                    base + (size_t)p * PROJ, D, D);
      transpose_cvt_bf16<<<gDF, blk, 0, stream>>>(W1 + (size_t)l * FFE, base + 8 * PROJ, D, FF);
      transpose_cvt_bf16<<<gFD, blk, 0, stream>>>(W2 + (size_t)l * FFE, base + 8 * PROJ + FFE,
                                                  FF, D);
    }
  }

  embed_pe<<<(M * D) / 256, 256, 0, stream>>>(dec, emb, x, T);

  const dim3 gP(D / 128, M / 128);
  const dim3 gF1(FF / 128, M / 128);
  const dim3 gA(T / 128, H, B);
  const dim3 gLN(M / 8);

  for (int l = 0; l < L; ++l) {
    const bf16_t* wb = Wt + (size_t)l * LSTR;
    const bf16_t* wqs = wb + 0 * PROJ, *wks = wb + 1 * PROJ, *wvs = wb + 2 * PROJ;
    const bf16_t* wos = wb + 3 * PROJ, *wqc = wb + 4 * PROJ, *wkc = wb + 5 * PROJ;
    const bf16_t* wvc = wb + 6 * PROJ, *woc = wb + 7 * PROJ;
    const bf16_t* w1t = wb + 8 * PROJ;        // [FF][D]
    const bf16_t* w2t = wb + 8 * PROJ + FFE;  // [D][FF]

    // ---- self attention ----
    gemm_bf16_wmma<<<gP, 256, 0, stream>>>(x, wqs, nullptr, Qb, M, D, D, 0);
    gemm_bf16_wmma<<<gP, 256, 0, stream>>>(x, wks, nullptr, Kb, M, D, D, 0);
    gemm_bf16_wmma<<<gP, 256, 0, stream>>>(x, wvs, nullptr, Vb, M, D, D, 0);
    attn_wmma<<<gA, 256, 0, stream>>>(Qb, Kb, Vb, dec, Cb, T, S, 1);
    gemm_bf16_wmma<<<gP, 256, 0, stream>>>(Cb, wos, x, x, M, D, D, 0);
    layernorm_rows<<<gLN, 256, 0, stream>>>(x, M);

    // ---- cross attention ----
    gemm_bf16_wmma<<<gP, 256, 0, stream>>>(x, wqc, nullptr, Qb, M, D, D, 0);
    gemm_bf16_wmma<<<gP, 256, 0, stream>>>(enc_out, wkc, nullptr, Kb, M, D, D, 0);
    gemm_bf16_wmma<<<gP, 256, 0, stream>>>(enc_out, wvc, nullptr, Vb, M, D, D, 0);
    attn_wmma<<<gA, 256, 0, stream>>>(Qb, Kb, Vb, enc, Cb, T, S, 0);
    gemm_bf16_wmma<<<gP, 256, 0, stream>>>(Cb, woc, x, x, M, D, D, 0);
    layernorm_rows<<<gLN, 256, 0, stream>>>(x, M);

    // ---- FFN ----
    gemm_bf16_wmma<<<gF1, 256, 0, stream>>>(x, w1t, nullptr, Hb, M, FF, D, 1);
    gemm_bf16_wmma<<<gP, 256, 0, stream>>>(Hb, w2t, x, x, M, D, FF, 0);
    layernorm_rows<<<gLN, 256, 0, stream>>>(x, M);
  }
}
